// Hankel_22746146800318
// MI455X (gfx1250) — compile-verified, matrix operand order checked by reading, not run on hardware
//
#include <hip/hip_runtime.h>

// MI455X / gfx1250 (CDNA5, wave32). Matrix math on v_wmma_f32_16x16x32_bf16.
// pack_b_kernel: fp32 weights -> bf16 WMMA B-fragment layout (one 32B load per
// lane per fragment). mps_chain_kernel: fused encoder + MPS chain, 16 batch
// rows per wave, 2 waves per block (256 blocks for whole-chip coverage).
// H_mid fragments are staged per 4-p group into a double-buffered 2x64KB LDS
// stage with the Tensor Data Mover (tensor_load_to_lds, in-order TENSORcnt:
// s_wait_tensorcnt 1 overlaps the next DMA with current compute). U-fragment
// scaling uses v_pk_mul_bf16.

typedef __attribute__((ext_vector_type(16))) __bf16 v16bf;
typedef __attribute__((ext_vector_type(8)))  __bf16 v8bf;
typedef __attribute__((ext_vector_type(8)))  float  v8f;
typedef __attribute__((ext_vector_type(4)))  float  v4f;
typedef __attribute__((ext_vector_type(4)))  unsigned int u32x4;
typedef __attribute__((ext_vector_type(8)))  int i32x8;
typedef __attribute__((ext_vector_type(4)))  int i32x4;

#define B_TRAJ 8192
#define T_LEN  12
#define DIN    64
#define WAVES  2                                  // waves per block

// LDS carve (dynamic shared memory)
#define STAGE_BYTES  65536                        // 4 p x 4 ec x 4 nt x 32 x 16 bf16
#define NBUF         2
#define ESTAGE_OFF   (STAGE_BYTES * NBUF)         // WAVES x 16 x 136 bf16
#define VBUF_OFF     (ESTAGE_OFF + WAVES*16*136*2)// WAVES x 16 x 68 f32
#define HSTAGE_OFF   (VBUF_OFF + WAVES*16*68*4)   // WAVES x 16 x 40 bf16
#define SMEM_BYTES   (HSTAGE_OFF + WAVES*16*40*2)

__device__ __forceinline__ v16bf concat8(v8bf a, v8bf b) {
  return __builtin_shufflevector(a, b, 0,1,2,3,4,5,6,7,8,9,10,11,12,13,14,15);
}

__device__ __forceinline__ v8bf cvt8(v8f f) {
  v8bf r;
#pragma unroll
  for (int i = 0; i < 8; ++i) r[i] = (__bf16)f[i];
  return r;
}

__device__ __forceinline__ v8f wmma_bf16(v16bf a, v16bf b, v8f c) {
  return __builtin_amdgcn_wmma_f32_16x16x32_bf16(false, a, false, b, (short)0, c,
                                                 false, false);
}

union bfvec16 { v16bf bf; unsigned u[8]; };

// au = ae * broadcast(vs) in packed bf16 (v_pk_mul_bf16, VOP3P)
__device__ __forceinline__ v16bf scale_frag_pk(v16bf ae, unsigned vsp) {
  bfvec16 a, r;
  a.bf = ae;
#pragma unroll
  for (int i = 0; i < 8; ++i)
    asm("v_pk_mul_bf16 %0, %1, %2" : "=v"(r.u[i]) : "v"(a.u[i]), "v"(vsp));
  return r.bf;
}

// ---- Tensor Data Mover: 1-D contiguous global -> LDS DMA (issue only) ------
// D# per CDNA5 ISA ch.8: group0 {count=1, lds_addr, global_addr, type=2},
// group1 {data_size=4B, tensor_dim0=tile_dim0=ndwords, tensor_dim1=1}.
// 6-arg builtin on this toolchain: (g0, g1, g2, g3, g4, cpol).
__device__ __forceinline__ void tdm_issue_1d(const void* gptr, unsigned lds_off,
                                             unsigned ndwords) {
#if __has_builtin(__builtin_amdgcn_tensor_load_to_lds)
  unsigned long long ga = (unsigned long long)(uintptr_t)gptr;
  u32x4 g0;
  g0[0] = 1u;                                          // count=1, user D#
  g0[1] = lds_off;                                     // lds_addr (bytes)
  g0[2] = (unsigned)(ga & 0xFFFFFFFFu);                // global_addr[31:0]
  g0[3] = (unsigned)((ga >> 32) & 0x01FFFFFFu) | 0x80000000u; // [56:32] | type=2
  i32x8 g1;
  g1[0] = 0x00020000;                                  // data_size = 4 bytes
  g1[1] = (int)(ndwords << 16);                        // tensor_dim0[15:0]
  g1[2] = (int)((ndwords >> 16) | (1u << 16));         // dim0 hi | tensor_dim1=1
  g1[3] = (int)(ndwords << 16);                        // tile_dim0 (16-bit)
  g1[4] = 0;                                           // tile_dim1/2 = 0 (1-D)
  g1[5] = (int)ndwords;                                // tensor_dim0_stride
  g1[6] = 0;
  g1[7] = 0;
  i32x4 gz4 = {0, 0, 0, 0};
  i32x8 gz8 = {0, 0, 0, 0, 0, 0, 0, 0};
  __builtin_amdgcn_tensor_load_to_lds(g0, g1, gz4, gz4, gz8, 0);
#else
  // Fallback: synchronous cooperative copy by the issuing wave (32 lanes)
  const v16bf* src = (const v16bf*)gptr;
  extern __shared__ char smem_fb[];
  v16bf* dst = (v16bf*)(smem_fb + lds_off);
  int chunks = (int)(ndwords >> 3);                    // 32B chunks
  for (int i = (int)(threadIdx.x & 31); i < chunks; i += 32) dst[i] = src[i];
#endif
}

__device__ __forceinline__ void tdm_wait_le1() {
#if __has_builtin(__builtin_amdgcn_tensor_load_to_lds)
#if __has_builtin(__builtin_amdgcn_s_wait_tensorcnt)
  __builtin_amdgcn_s_wait_tensorcnt(1);
#else
  asm volatile("s_wait_tensorcnt 0x1" ::: "memory");
#endif
#endif
}

__device__ __forceinline__ void tdm_wait0() {
#if __has_builtin(__builtin_amdgcn_tensor_load_to_lds)
#if __has_builtin(__builtin_amdgcn_s_wait_tensorcnt)
  __builtin_amdgcn_s_wait_tensorcnt(0);
#else
  asm volatile("s_wait_tensorcnt 0x0" ::: "memory");
#endif
#endif
}

// Pack an fp32 matrix into bf16 WMMA B-fragments (ISA 16-bit B layout):
//   lanes 0-15 : N=lane,    K = kc*32 + {0..7, 16..23}
//   lanes 16-31: N=lane-16, K = kc*32 + {8..15, 24..31}
__global__ void pack_b_kernel(const float* __restrict__ src,
                              __bf16* __restrict__ dst,
                              int kchunks, int ntiles, int nstride, int transposed) {
  int tid = blockIdx.x * blockDim.x + threadIdx.x;
  int total = kchunks * ntiles * 32;
  if (tid >= total) return;
  int lane = tid & 31;
  int frag = tid >> 5;
  int kc = frag / ntiles;
  int nt = frag - kc * ntiles;
  int n = nt * 16 + (lane & 15);
  int kbase = kc * 32 + ((lane >> 4) ? 8 : 0);
  __bf16* out = dst + (size_t)tid * 16;
#pragma unroll
  for (int j = 0; j < 16; ++j) {
    int k = kbase + (j < 8 ? j : j + 8);
    float v = transposed ? src[(size_t)n * nstride + k]
                         : src[(size_t)k * nstride + n];
    out[j] = (__bf16)v;
  }
}

__global__ __launch_bounds__(64, 1)
void mps_chain_kernel(const float* __restrict__ x,
                      const float* __restrict__ b1,
                      const float* __restrict__ b2,
                      const __bf16* __restrict__ W1p,   // [2 kc][32 nt] frags
                      const __bf16* __restrict__ W2p,   // [16 kc][8 nt]
                      const __bf16* __restrict__ Hfp,   // [4 kc][4 nt]
                      const __bf16* __restrict__ Hlp,   // [4 kc][4 nt]
                      const __bf16* __restrict__ Hmp,   // [10 t][64 p][4 ec][4 nt]
                      float* __restrict__ out) {
  extern __shared__ char smem[];
  __bf16* stage  = (__bf16*)smem;                  // 2 x 64KB H_mid stage
  __bf16* estage = (__bf16*)(smem + ESTAGE_OFF);   // per-wave enc tile 16x128
  float*  vbuf   = (float*)(smem + VBUF_OFF);      // per-wave chain state 16x64
  __bf16* hstage = (__bf16*)(smem + HSTAGE_OFF);   // per-wave 16x32 h slice

  const int lane = threadIdx.x & 31;
  const int wave = threadIdx.x >> 5;
  const int half = lane >> 4;
  const int r16  = lane & 15;
  const int b0   = (blockIdx.x * WAVES + wave) * 16;
  const unsigned stage_lds0 = (unsigned)(uintptr_t)stage;

  __bf16* est = estage + (size_t)wave * 16 * 136;
  float*  vb  = vbuf   + (size_t)wave * 16 * 68;
  __bf16* hst = hstage + (size_t)wave * 16 * 40;

  for (int t = 0; t < T_LEN; ++t) {
    // ---------------- encoder: A-fragments of x (fp32 -> bf16 inline) -----
    v16bf ax[2];
#pragma unroll
    for (int kc = 0; kc < 2; ++kc) {
      const float* p = x + ((size_t)(b0 + r16) * T_LEN + t) * DIN + kc * 32 + half * 8;
      v8f lo = *(const v8f*)p;
      v8f hi = *(const v8f*)(p + 16);
      ax[kc] = concat8(cvt8(lo), cvt8(hi));
    }

    v8f acc2[8];
#pragma unroll
    for (int nt = 0; nt < 8; ++nt) {
      float bv = b2[nt * 16 + r16];
#pragma unroll
      for (int i = 0; i < 8; ++i) acc2[nt][i] = bv;
    }

    for (int kt = 0; kt < 16; ++kt) {   // 16 K-chunks of 32 over HID=512
#pragma unroll
      for (int j = 0; j < 2; ++j) {
        int nt1 = kt * 2 + j;
        float bv = b1[nt1 * 16 + r16];
        v8f acc1;
#pragma unroll
        for (int i = 0; i < 8; ++i) acc1[i] = bv;
#pragma unroll
        for (int kc = 0; kc < 2; ++kc) {
          v16bf bf = *(const v16bf*)(W1p + (((size_t)kc * 32 + nt1) * 32 + lane) * 16);
          acc1 = wmma_bf16(ax[kc], bf, acc1);
        }
#pragma unroll
        for (int i = 0; i < 8; ++i)
          hst[(i + 8 * half) * 40 + j * 16 + r16] = (__bf16)fmaxf(acc1[i], 0.f);
      }
      __syncthreads();
      const __bf16* hrow = hst + r16 * 40 + half * 8;
      v16bf ah = concat8(*(const v8bf*)hrow, *(const v8bf*)(hrow + 16));
#pragma unroll
      for (int nt = 0; nt < 8; ++nt) {
        v16bf bf = *(const v16bf*)(W2p + (((size_t)kt * 8 + nt) * 32 + lane) * 16);
        acc2[nt] = wmma_bf16(ah, bf, acc2[nt]);
      }
      __syncthreads();
    }

#pragma unroll
    for (int nt = 0; nt < 8; ++nt)
#pragma unroll
      for (int i = 0; i < 8; ++i)
        est[(i + 8 * half) * 136 + nt * 16 + r16] = (__bf16)fmaxf(acc2[nt][i], 0.f);
    __syncthreads();

    v16bf ae[4];
#pragma unroll
    for (int kc = 0; kc < 4; ++kc) {
      const __bf16* erow = est + r16 * 136 + kc * 32 + half * 8;
      ae[kc] = concat8(*(const v8bf*)erow, *(const v8bf*)(erow + 16));
    }

    if (t == 0) {
      v8f accv[4];
#pragma unroll
      for (int nt = 0; nt < 4; ++nt)
#pragma unroll
        for (int i = 0; i < 8; ++i) accv[nt][i] = 0.f;
#pragma unroll
      for (int kc = 0; kc < 4; ++kc)
#pragma unroll
        for (int nt = 0; nt < 4; ++nt) {
          v16bf bf = *(const v16bf*)(Hfp + (((size_t)kc * 4 + nt) * 32 + lane) * 16);
          accv[nt] = wmma_bf16(ae[kc], bf, accv[nt]);
        }
#pragma unroll
      for (int nt = 0; nt < 4; ++nt)
#pragma unroll
        for (int i = 0; i < 8; ++i)
          vb[(i + 8 * half) * 68 + nt * 16 + r16] = accv[nt][i];
    } else if (t < T_LEN - 1) {
      // v' = U @ H_mid[t-1] with U[b, p*128+e] = v[b,p]*enc[b,e]  (K = 8192)
      v8f accv[4];
#pragma unroll
      for (int nt = 0; nt < 4; ++nt)
#pragma unroll
        for (int i = 0; i < 8; ++i) accv[nt][i] = 0.f;
      const __bf16* Hbase = Hmp + (size_t)(t - 1) * (64 * 4 * 4 * 32 * 16);
      // prologue: DMA group 0 into buffer 0
      if (wave == 0)
        tdm_issue_1d(Hbase, stage_lds0, STAGE_BYTES / 4);
      for (int pg = 0; pg < 16; ++pg) {              // 4 p per group
        if (wave == 0) {                             // double-buffered TDM
          if (pg < 15) {
            tdm_issue_1d(Hbase + (size_t)(pg + 1) * (4 * 4 * 4 * 32 * 16),
                         stage_lds0 + ((pg + 1) & 1) * STAGE_BYTES,
                         STAGE_BYTES / 4);
            tdm_wait_le1();                          // oldest (pg) complete
          } else {
            tdm_wait0();
          }
        }
        __syncthreads();
        const __bf16* sbuf = stage + (size_t)(pg & 1) * (STAGE_BYTES / 2);
        v4f vs4 = *(const v4f*)(vb + r16 * 68 + pg * 4);
#pragma unroll
        for (int pl = 0; pl < 4; ++pl) {
          float vs = vs4[pl];
          unsigned hb = (unsigned)__builtin_bit_cast(unsigned short, (__bf16)vs);
          unsigned vsp = hb | (hb << 16);
#pragma unroll
          for (int ec = 0; ec < 4; ++ec) {
            v16bf au = scale_frag_pk(ae[ec], vsp);
#pragma unroll
            for (int nt = 0; nt < 4; ++nt) {
              v16bf bf = *(const v16bf*)(sbuf +
                           ((((size_t)pl * 4 + ec) * 4 + nt) * 32 + lane) * 16);
              accv[nt] = wmma_bf16(au, bf, accv[nt]);
            }
          }
        }
        __syncthreads();   // all readers done before this buffer is re-targeted
      }
#pragma unroll
      for (int nt = 0; nt < 4; ++nt)
#pragma unroll
        for (int i = 0; i < 8; ++i)
          vb[(i + 8 * half) * 68 + nt * 16 + r16] = accv[nt][i];
    } else {
      v8f accl[4];
#pragma unroll
      for (int nt = 0; nt < 4; ++nt)
#pragma unroll
        for (int i = 0; i < 8; ++i) accl[nt][i] = 0.f;
#pragma unroll
      for (int kc = 0; kc < 4; ++kc)
#pragma unroll
        for (int nt = 0; nt < 4; ++nt) {
          v16bf bf = *(const v16bf*)(Hlp + (((size_t)kc * 4 + nt) * 32 + lane) * 16);
          accl[nt] = wmma_bf16(ae[kc], bf, accl[nt]);
        }
      float si[8];
#pragma unroll
      for (int i = 0; i < 8; ++i) si[i] = 0.f;
#pragma unroll
      for (int nt = 0; nt < 4; ++nt)
#pragma unroll
        for (int i = 0; i < 8; ++i)
          si[i] += accl[nt][i] * vb[(i + 8 * half) * 68 + nt * 16 + r16];
#pragma unroll
      for (int i = 0; i < 8; ++i) {
#pragma unroll
        for (int m = 8; m >= 1; m >>= 1) si[i] += __shfl_xor(si[i], m, 32);
      }
      if (r16 == 0) {
#pragma unroll
        for (int i = 0; i < 8; ++i) out[b0 + 8 * half + i] = si[i];
      }
    }
    __syncthreads();
  }
}

extern "C" void kernel_launch(void* const* d_in, const int* in_sizes, int n_in,
                              void* d_out, int out_size, void* d_ws, size_t ws_size,
                              hipStream_t stream) {
  (void)in_sizes; (void)n_in; (void)out_size; (void)ws_size;
  const float* x  = (const float*)d_in[0];   // [8192,12,64]
  const float* W1 = (const float*)d_in[1];   // [512,64]
  const float* b1 = (const float*)d_in[2];   // [512]
  const float* W2 = (const float*)d_in[3];   // [128,512]
  const float* b2 = (const float*)d_in[4];   // [128]
  const float* Hf = (const float*)d_in[5];   // [1,128,64]
  const float* Hm = (const float*)d_in[6];   // [10,64,128,64]
  const float* Hl = (const float*)d_in[7];   // [64,128,1]
  float* out = (float*)d_out;

  char* ws = (char*)d_ws;
  size_t off = 0;
  auto take = [&](size_t elems) -> __bf16* {
    __bf16* p = (__bf16*)(ws + off);
    off = (off + elems * sizeof(__bf16) + 255) & ~(size_t)255;
    return p;
  };
  __bf16* W1p = take((size_t)2 * 32 * 32 * 16);      //  64 KB
  __bf16* W2p = take((size_t)16 * 8 * 32 * 16);      // 128 KB
  __bf16* Hfp = take((size_t)4 * 4 * 32 * 16);       //  16 KB
  __bf16* Hlp = take((size_t)4 * 4 * 32 * 16);       //  16 KB
  __bf16* Hmp = take((size_t)2560 * 4 * 32 * 16);    //  10 MB

  auto pack = [&](const float* src, __bf16* dst, int kc, int nt, int ns, int tr) {
    int total = kc * nt * 32;
    pack_b_kernel<<<(total + 255) / 256, 256, 0, stream>>>(src, dst, kc, nt, ns, tr);
  };
  pack(W1, W1p, 2, 32, 64, 1);        // B[d][h]  = W1[h][d]
  pack(W2, W2p, 16, 8, 512, 1);       // B[h][e]  = W2[e][h]
  pack(Hf, Hfp, 4, 4, 64, 0);         // B[e][r]  = Hf[e][r]
  pack(Hl, Hlp, 4, 4, 128, 1);        // B[e][p]  = Hl[p][e]
  pack(Hm, Hmp, 2560, 4, 64, 0);      // B[pe][r] = Hm flat, 10 x 8192 x 64

  mps_chain_kernel<<<B_TRAJ / (16 * WAVES), 64, SMEM_BYTES, stream>>>(
      x, b1, b2, W1p, W2p, Hfp, Hlp, Hmp, out);
}
